// KTBM_43404939494172
// MI455X (gfx1250) — compile-verified, matrix operand order checked by reading, not run on hardware
//
#include <hip/hip_runtime.h>

typedef _Float16 v16h __attribute__((ext_vector_type(16)));
typedef float v8f __attribute__((ext_vector_type(8)));

#define NWAVES 4
#define TPB 128

struct P { const float* w[66]; };

// ---------------- packed-weight descriptor table ----------------
struct MDesc { int widx, K, N, wT; };

__host__ __device__ constexpr MDesc MD[] = {
  {12,  64,  64, 0},  // 0  MQ   mapQ_key_W
  {14,  64,  64, 0},  // 1  ML   mapL_key_W
  {10,  64,  32, 1},  // 2  KEYT key_matrix^T (stored 32x64)
  {16, 128,  64, 0},  // 3  EQ   erase_Q_W
  {18,  64,  64, 0},  // 4  EL   erase_L_W
  {20, 128,  64, 0},  // 5  EBH  erase_bh_W
  {21, 128,  64, 0},  // 6  AQ   add_Q_W
  {23,  64,  64, 0},  // 7  AL   add_L_W
  {25, 128,  64, 0},  // 8  ABH  add_bh_W
  {26,  64,  64, 0},  // 9  TQQ
  {27,  64,  64, 0},  // 10 TQL
  {28,  64,  64, 0},  // 11 TLQ
  {29,  64,  64, 0},  // 12 TLL
  {30, 128, 128, 0},  // 13 SUMM
  {34,  96, 128, 0},  // 14 BMQ
  {36,  96, 128, 0},  // 15 BML
  {40, 224, 128, 0},  // 16 PRQ
  {42, 224, 128, 0},  // 17 PRL
  {46, 128, 128, 0},  // 18 IX
  {48, 128, 128, 0},  // 19 IH
  {49,  64, 128, 0},  // 20 IK
  {56, 128, 128, 0},  // 21 FX
  {58, 128, 128, 0},  // 22 FH
  {59,  64, 128, 0},  // 23 FK
  {51, 128, 128, 0},  // 24 GX
  {53, 128, 128, 0},  // 25 GH
  {54,  64, 128, 0},  // 26 GK
  {61, 128, 128, 0},  // 27 OX
  {63, 128, 128, 0},  // 28 OH
  {64,  64, 128, 0},  // 29 OK
};
enum { MQ=0, ML, KEYT, EQ, EL, EBH, AQ, AL, ABH, TQQ, TQL, TLQ, TLL,
       SUMM, BMQ, BML, PRQ, PRL, IX, IH, IK, FX, FH, FK, GX, GH, GK, OX, OH, OK,
       NMAT };

__host__ __device__ constexpr int mat_off(int m) {
  int o = 0;
  for (int i = 0; i < m; ++i) o += MD[i].K * MD[i].N;
  return o;
}
// vg: 8 blocks x 16 batch x (32x64) halves
constexpr int VWS_HALVES = 8 * 16 * 2048;

__device__ __forceinline__ int frag_k(int e, int hi8) {
  return (e < 8) ? (hi8 + e) : (16 + hi8 + (e - 8));
}

// ---------------- one-time weight pack kernel ----------------
__global__ __launch_bounds__(TPB) void pack_kernel(P p, _Float16* __restrict__ dst) {
  int m = blockIdx.x;
  MDesc d = MD[m];
  const float* __restrict__ W = p.w[d.widx];
  _Float16* __restrict__ o = dst + mat_off(m);
  int NT = d.N >> 4;
  int total = d.K * d.N;
  for (int idx = threadIdx.x; idx < total; idx += TPB) {
    int chunk = idx >> 9, rem = idx & 511;
    int lane_ = rem >> 4, e = rem & 15;
    int kt = chunk / NT, nt = chunk - kt * NT;
    int hi8 = (lane_ & 16) ? 8 : 0;
    int k = kt * 32 + frag_k(e, hi8);
    int n = nt * 16 + (lane_ & 15);
    float x = d.wT ? W[n * d.K + k] : W[k * d.N + n];
    o[idx] = (_Float16)x;
  }
}

// ---------------- GEMM helper ----------------
struct Term {
  const _Float16* A; int lda;    // 16-row f16 A operand (LDS or global)
  const _Float16* B;             // packed fragments (base already offset for kt0)
  int K;                         // reduction length (multiple of 32)
};

__device__ __forceinline__ float sigf(float x) { return 1.f / (1.f + __expf(-x)); }

__device__ __forceinline__ v16h load_a_frag(const _Float16* __restrict__ A, int lda,
                                            int kb, int lane) {
  v16h f;
  const _Float16* row = A + (lane & 15) * lda + kb;
  int hi8 = (lane & 16) ? 8 : 0;
#pragma unroll
  for (int e = 0; e < 8; ++e) f[e] = row[hi8 + e];          // 16B contiguous
#pragma unroll
  for (int e = 0; e < 8; ++e) f[8 + e] = row[16 + hi8 + e]; // 16B contiguous
  return f;
}

// MODE: 0 store ACT(x) to rows where mask bit == WANT (ACT 0/1/2 = id/sig/tanh)
//       1 out = sig(x)              (i gate)
//       2 out = sig(x)*out          (f gate: m *= f)
//       3 out += aux*tanh(x)        (g gate: m += i*g)
//       4 out = sig(x)*tanh(aux)    (o gate: h_new)
template <int NTERMS, int N, int MODE, int ACT, bool BIAS1, bool BIAS2, bool RSEL,
          bool OUT16>
__device__ __forceinline__ void gemm16(const Term (&t)[NTERMS],
                                       const float* __restrict__ bias,
                                       const float* __restrict__ bias2,
                                       float* out, _Float16* out16, int ldo,
                                       float* aux, int ldaux,
                                       unsigned rmask, int want, int wvid, int lane) {
  constexpr int NT = N >> 4;
#pragma unroll
  for (int nt = wvid; nt < NT; nt += NWAVES) {
    int n0 = nt << 4;
    v8f acc = {};
#pragma unroll
    for (int ti = 0; ti < NTERMS; ++ti) {
      const _Float16* A = t[ti].A;
      int lda = t[ti].lda;
      const _Float16* B = t[ti].B;
      int KT = t[ti].K >> 5;
      for (int kk = 0; kk < KT; ++kk) {
        v16h a = load_a_frag(A, lda, kk << 5, lane);
        v16h b = *(const v16h*)(B + (((kk * NT + nt) << 9) + (lane << 4)));
        acc = __builtin_amdgcn_wmma_f32_16x16x32_f16(false, a, false, b, (short)0, acc,
                                                     false, false);
      }
    }
    int nn = n0 + (lane & 15);
    int rb = (lane & 16) ? 8 : 0;
    float bv = 0.f;
    if (BIAS1) bv += bias[nn];
    if (BIAS2) bv += bias2[nn];
#pragma unroll
    for (int i = 0; i < 8; ++i) {
      int r = rb + i;
      float x = acc[i] + bv;
      if (MODE == 0) {
        float y = (ACT == 1) ? sigf(x) : (ACT == 2) ? tanhf(x) : x;
        bool wr = !RSEL || (((rmask >> r) & 1u) == (unsigned)want);
        if (wr) {
          if (OUT16) out16[r * ldo + nn] = (_Float16)y;
          else       out[r * ldo + nn] = y;
        }
      } else if (MODE == 1) {
        out[r * ldo + nn] = sigf(x);
      } else if (MODE == 2) {
        out[r * ldo + nn] = sigf(x) * out[r * ldo + nn];
      } else if (MODE == 3) {
        out[r * ldo + nn] += aux[r * ldaux + nn] * tanhf(x);
      } else {
        out[r * ldo + nn] = sigf(x) * tanhf(aux[r * ldaux + nn]);
      }
    }
  }
}

__device__ __forceinline__ void softmax32(float* buf, int tid) {
  if (tid < 16) {
    float* row = buf + tid * 32;
    float mx = row[0];
    for (int c = 1; c < 32; ++c) mx = fmaxf(mx, row[c]);
    float s = 0.f;
    for (int c = 0; c < 32; ++c) { float e = __expf(row[c] - mx); row[c] = e; s += e; }
    float inv = 1.f / s;
    for (int c = 0; c < 32; ++c) row[c] *= inv;
  }
}

// f32 shared (floats)
enum {
  F_M   = 0,     // 16x128 persistent
  F_W   = 2048,  // 16x32  | phase4: RCW
  F_E   = 2560,  // 16x64  | phase3: SI (2048, covers E+AV)
  F_AV  = 3584,  // 16x64
  F_VT  = 4608,  // 16x64
  F_KVA = 5632,  // 16x64 kvec raw accumulator
  F_PS  = 6656,  // 16x128 PREF then SUMM
  SMF   = 8704
};
#define F_SI  F_E
// f16 shared (halves)
enum {
  H_QE  = 0,
  H_AE  = 1024,
  H_LE  = 2048,
  H_DE  = 3072,  // 16x32
  H_QBE = 3584,
  H_LBE = 4608,
  H_H   = 5632,  // 16x128 persistent
  H_XB  = 7680,  // 16x128
  H_KV  = 9728,  // 16x64
  H_KEY = 10752, // 16x64 | phase4: QNK
  H_RC  = 11776, // 16x64
  H_QN  = 12800, // 16x64
  SMH   = 13824
};

__global__ __launch_bounds__(TPB) void ktbm_kernel(
    const int* __restrict__ qd, const int* __restrict__ ad,
    const int* __restrict__ lddat, const int* __restrict__ dd,
    P p, float* __restrict__ out, _Float16* __restrict__ vgbase,
    const _Float16* __restrict__ pk) {
  __shared__ float smf[SMF];
  __shared__ _Float16 smh[SMH];
  __shared__ int sq[16], sa[16], sl[16], sdt[16], sdp[16];
  __shared__ unsigned s_dtmask;

  const int tid = threadIdx.x;
  const int wvid = tid >> 5;
  const int lane = tid & 31;
  const int b0 = blockIdx.x * 16;
  const int S = 200, SO = S - 2;

  _Float16* vg = vgbase + (size_t)blockIdx.x * 16 * 2048;

  const float* vinit = p.w[11];
  for (int i = tid; i < 16 * 2048; i += TPB) vg[i] = (_Float16)vinit[i & 2047];
  for (int i = tid; i < 2048; i += TPB) {
    smf[F_M + i] = 0.f;
    smh[H_H + i] = (_Float16)0.f;
  }
  __syncthreads();

  for (int t = 1; t <= S - 2; ++t) {
    if (tid < 16) {
      int b = b0 + tid;
      sq[tid]  = qd[b * S + t];
      sa[tid]  = ad[b * S + t];
      sl[tid]  = lddat[b * S + t];
      sdt[tid] = dd[b * S + t];
      sdp[tid] = dd[b * S + t - 1];
    }
    __syncthreads();
    if (tid == 0) {
      unsigned m = 0;
      for (int r = 0; r < 16; ++r) m |= (unsigned)(sdt[r] & 1) << r;
      s_dtmask = m;
    }
    for (int i = tid; i < 1024; i += TPB) {
      int r = i >> 6, c = i & 63;
      smh[H_QE  + i] = (_Float16)p.w[4][sq[r] * 64 + c];
      smh[H_AE  + i] = (_Float16)p.w[5][sa[r] * 64 + c];
      smh[H_LE  + i] = (_Float16)p.w[6][sl[r] * 64 + c];
      smh[H_QBE + i] = (_Float16)p.w[8][sq[r] * 64 + c];
      smh[H_LBE + i] = (_Float16)p.w[9][sl[r] * 64 + c];
    }
    for (int i = tid; i < 512; i += TPB) {
      int r = i >> 5, c = i & 31;
      smh[H_DE + i] = (_Float16)p.w[7][sdt[r] * 32 + c];
    }
    __syncthreads();
    const unsigned dtm = s_dtmask;

    // ---- attention key (row-select by d_t), logits, softmax -> w ----
    {
      Term tq[1] = {{smh + H_QE, 64, pk + mat_off(MQ), 64}};
      gemm16<1, 64, 0, 0, true, false, true, true>(tq, p.w[13], nullptr, nullptr,
          smh + H_KEY, 64, nullptr, 0, dtm, 0, wvid, lane);
      Term tl[1] = {{smh + H_LE, 64, pk + mat_off(ML), 64}};
      gemm16<1, 64, 0, 0, true, false, true, true>(tl, p.w[15], nullptr, nullptr,
          smh + H_KEY, 64, nullptr, 0, dtm, 1, wvid, lane);
    }
    __syncthreads();
    {
      Term tk[1] = {{smh + H_KEY, 64, pk + mat_off(KEYT), 64}};
      gemm16<1, 32, 0, 0, false, false, false, false>(tk, nullptr, nullptr,
          smf + F_W, nullptr, 32, nullptr, 0, 0u, 0, wvid, lane);
    }
    __syncthreads();
    softmax32(smf + F_W, tid);
    __syncthreads();

    // ---- erase / add ----
    {
      Term te0[3] = {{smh + H_H, 128, pk + mat_off(EBH), 128},
                     {smh + H_QE, 64, pk + mat_off(EQ), 64},
                     {smh + H_AE, 64, pk + mat_off(EQ) + 2 * 4 * 512, 64}};
      gemm16<3, 64, 0, 1, true, false, true, false>(te0, p.w[17], nullptr,
          smf + F_E, nullptr, 64, nullptr, 0, dtm, 0, wvid, lane);
      Term te1[2] = {{smh + H_H, 128, pk + mat_off(EBH), 128},
                     {smh + H_LE, 64, pk + mat_off(EL), 64}};
      gemm16<2, 64, 0, 1, true, false, true, false>(te1, p.w[19], nullptr,
          smf + F_E, nullptr, 64, nullptr, 0, dtm, 1, wvid, lane);
      Term ta0[3] = {{smh + H_H, 128, pk + mat_off(ABH), 128},
                     {smh + H_QE, 64, pk + mat_off(AQ), 64},
                     {smh + H_AE, 64, pk + mat_off(AQ) + 2 * 4 * 512, 64}};
      gemm16<3, 64, 0, 2, true, false, true, false>(ta0, p.w[22], nullptr,
          smf + F_AV, nullptr, 64, nullptr, 0, dtm, 0, wvid, lane);
      Term ta1[2] = {{smh + H_H, 128, pk + mat_off(ABH), 128},
                     {smh + H_LE, 64, pk + mat_off(AL), 64}};
      gemm16<2, 64, 0, 2, true, false, true, false>(ta1, p.w[24], nullptr,
          smf + F_AV, nullptr, 64, nullptr, 0, dtm, 1, wvid, lane);
    }
    for (int i = tid; i < 1024; i += TPB) smf[F_KVA + i] = 0.f;
    __syncthreads();

    // ---- value memory: vt = tanh(v @ T_sel), v_new; kvec accumulate ----
    for (int b = 0; b < 16; ++b) {
      const _Float16* Tsel = pk + mat_off(TQQ) + (sdp[b] * 2 + sdt[b]) * 4096;
      const float* sumkW = p.w[38];
      for (int h2 = 0; h2 < 2; ++h2) {
        _Float16* vb = vg + b * 2048 + h2 * 1024;
        Term tv[1] = {{vb, 64, Tsel, 64}};
        gemm16<1, 64, 0, 2, false, false, false, false>(tv, nullptr, nullptr,
            smf + F_VT, nullptr, 64, nullptr, 0, 0u, 0, wvid, lane);
        __syncthreads();
        for (int i = tid; i < 1024; i += TPB) {
          int c = h2 * 16 + (i >> 6), j = i & 63;
          float ww = smf[F_W + b * 32 + c];
          float vn = smf[F_VT + i] * (1.f - ww * smf[F_E + b * 64 + j]) +
                     ww * smf[F_AV + b * 64 + j];
          vb[i] = (_Float16)vn;
          atomicAdd(&smf[F_KVA + b * 64 + j], sumkW[c] * vn);
        }
        __syncthreads();
      }
    }
    {
      float skb = p.w[39][0];
      for (int i = tid; i < 1024; i += TPB)
        smh[H_KV + i] = (_Float16)tanhf(smf[F_KVA + i] + skb);
    }
    __syncthreads();

    // ---- behavior LSTM ----
    {
      Term txq[2] = {{smh + H_QBE, 64, pk + mat_off(BMQ), 64},
                     {smh + H_DE, 32, pk + mat_off(BMQ) + 2 * 8 * 512, 32}};
      gemm16<2, 128, 0, 0, true, false, true, true>(txq, p.w[35], nullptr, nullptr,
          smh + H_XB, 128, nullptr, 0, dtm, 0, wvid, lane);
      Term txl[2] = {{smh + H_LBE, 64, pk + mat_off(BML), 64},
                     {smh + H_DE, 32, pk + mat_off(BML) + 2 * 8 * 512, 32}};
      gemm16<2, 128, 0, 0, true, false, true, true>(txl, p.w[37], nullptr, nullptr,
          smh + H_XB, 128, nullptr, 0, dtm, 1, wvid, lane);
    }
    __syncthreads();
    {
      Term ti_[3] = {{smh + H_XB, 128, pk + mat_off(IX), 128},
                     {smh + H_H, 128, pk + mat_off(IH), 128},
                     {smh + H_KV, 64, pk + mat_off(IK), 64}};
      gemm16<3, 128, 1, 0, true, true, false, false>(ti_, p.w[47], p.w[50],
          smf + F_SI, nullptr, 128, nullptr, 0, 0u, 0, wvid, lane);
      Term tf_[3] = {{smh + H_XB, 128, pk + mat_off(FX), 128},
                     {smh + H_H, 128, pk + mat_off(FH), 128},
                     {smh + H_KV, 64, pk + mat_off(FK), 64}};
      gemm16<3, 128, 2, 0, true, true, false, false>(tf_, p.w[57], p.w[60],
          smf + F_M, nullptr, 128, nullptr, 0, 0u, 0, wvid, lane);
      Term tg_[3] = {{smh + H_XB, 128, pk + mat_off(GX), 128},
                     {smh + H_H, 128, pk + mat_off(GH), 128},
                     {smh + H_KV, 64, pk + mat_off(GK), 64}};
      gemm16<3, 128, 3, 0, true, true, false, false>(tg_, p.w[52], p.w[55],
          smf + F_M, nullptr, 128, smf + F_SI, 128, 0u, 0, wvid, lane);
      Term to_[3] = {{smh + H_XB, 128, pk + mat_off(OX), 128},
                     {smh + H_H, 128, pk + mat_off(OH), 128},
                     {smh + H_KV, 64, pk + mat_off(OK), 64}};
      gemm16<3, 128, 4, 0, true, true, false, false>(to_, p.w[62], p.w[65],
          smf + F_SI, nullptr, 128, smf + F_M, 128, 0u, 0, wvid, lane);
    }
    __syncthreads();
    for (int i = tid; i < 2048; i += TPB) smh[H_H + i] = (_Float16)smf[F_SI + i];
    __syncthreads();

    // ---- pref / pred_type ----
    if (tid < 16) sq[tid] = qd[(b0 + tid) * S + t + 1];
    __syncthreads();
    for (int i = tid; i < 1024; i += TPB) {
      int r = i >> 6, c = i & 63;
      smh[H_QN + i] = (_Float16)p.w[4][sq[r] * 64 + c];
    }
    __syncthreads();
    {
      Term tq[3] = {{smh + H_QBE, 64, pk + mat_off(PRQ), 64},
                    {smh + H_DE, 32, pk + mat_off(PRQ) + 2 * 8 * 512, 32},
                    {smh + H_H, 128, pk + mat_off(PRQ) + 3 * 8 * 512, 128}};
      gemm16<3, 128, 0, 0, true, false, true, false>(tq, p.w[41], nullptr,
          smf + F_PS, nullptr, 128, nullptr, 0, dtm, 0, wvid, lane);
      Term tl[3] = {{smh + H_LBE, 64, pk + mat_off(PRL), 64},
                    {smh + H_DE, 32, pk + mat_off(PRL) + 2 * 8 * 512, 32},
                    {smh + H_H, 128, pk + mat_off(PRL) + 3 * 8 * 512, 128}};
      gemm16<3, 128, 0, 0, true, false, true, false>(tl, p.w[43], nullptr,
          smf + F_PS, nullptr, 128, nullptr, 0, dtm, 1, wvid, lane);
    }
    __syncthreads();
    if (tid < 16) {
      float acc = p.w[45][0];
      for (int n = 0; n < 128; ++n) acc += smf[F_PS + tid * 128 + n] * p.w[44][n];
      out[128 * SO + (b0 + tid) * SO + (t - 1)] = sigf(acc);
    }
    {
      Term tk[1] = {{smh + H_QN, 64, pk + mat_off(MQ), 64}};
      gemm16<1, 64, 0, 0, true, false, false, true>(tk, p.w[13], nullptr, nullptr,
          smh + H_KEY, 64, nullptr, 0, 0u, 0, wvid, lane);
    }
    __syncthreads();
    {
      Term tk[1] = {{smh + H_KEY, 64, pk + mat_off(KEYT), 64}};
      gemm16<1, 32, 0, 0, false, false, false, false>(tk, nullptr, nullptr,
          smf + F_W, nullptr, 32, nullptr, 0, 0u, 0, wvid, lane);
    }
    __syncthreads();
    softmax32(smf + F_W, tid);
    __syncthreads();
    for (int i = tid; i < 1024; i += TPB) {  // rc = rcw . v_new over concepts
      int r = i >> 6, j = i & 63;
      float acc = 0.f;
      const _Float16* vb = vg + r * 2048;
      for (int c = 0; c < 32; ++c) acc += smf[F_W + r * 32 + c] * (float)vb[c * 64 + j];
      smh[H_RC + i] = (_Float16)acc;
    }
    __syncthreads();
    {
      Term ts[2] = {{smh + H_RC, 64, pk + mat_off(SUMM), 64},
                    {smh + H_QN, 64, pk + mat_off(SUMM) + 2 * 8 * 512, 64}};
      gemm16<2, 128, 0, 2, true, false, false, false>(ts, p.w[31], nullptr,
          smf + F_PS, nullptr, 128, nullptr, 0, 0u, 0, wvid, lane);
    }
    __syncthreads();
    if (tid < 16) {
      float acc = p.w[33][0];
      for (int n = 0; n < 128; ++n) acc += smf[F_PS + tid * 128 + n] * p.w[32][n];
      out[(b0 + tid) * SO + (t - 1)] = sigf(acc);
    }
    __syncthreads();
  }
}

extern "C" void kernel_launch(void* const* d_in, const int* in_sizes, int n_in,
                              void* d_out, int out_size, void* d_ws, size_t ws_size,
                              hipStream_t stream) {
  (void)in_sizes; (void)out_size; (void)ws_size;
  P p;
  for (int i = 0; i < 66; ++i) p.w[i] = (i < n_in) ? (const float*)d_in[i] : nullptr;
  _Float16* vg = reinterpret_cast<_Float16*>(d_ws);
  _Float16* pk = vg + VWS_HALVES;
  pack_kernel<<<NMAT, TPB, 0, stream>>>(p, pk);
  ktbm_kernel<<<8, TPB, 0, stream>>>(
      (const int*)d_in[0], (const int*)d_in[1], (const int*)d_in[2], (const int*)d_in[3],
      p, (float*)d_out, vg, pk);
}
